// Net_stacked_modified_87050397155867
// MI455X (gfx1250) — compile-verified, matrix operand order checked by reading, not run on hardware
//
#include <hip/hip_runtime.h>
#include <hip/hip_bf16.h>
#include <math.h>

// ---------------- problem constants ----------------
#define NB    2048          // batch
#define DD    256           // dim
#define HH    266           // hidden
#define SS    50            // steps
#define HP    272           // padded hidden cols (17*16)
#define KP    288           // padded K (9*32)
#define ROWS  32            // batch rows per block
#define NBLK  (NB/ROWS)     // 64 persistent blocks
#define NTHD  256           // 8 waves (wave32)
#define NEVT  (3 + 2*SS)    // BN events: v0 L1,L2,L3 + 2 per step
#define KAPPA 1.0f
#define SIGMA 0.3f
#define EPSBN 1e-5f

typedef __attribute__((ext_vector_type(16))) __bf16 v16bf;
typedef __attribute__((ext_vector_type(8)))  __bf16 v8bf;
typedef __attribute__((ext_vector_type(8)))  float  v8f;

// ---------------- async copy: global bf16 (pre-transposed) -> LDS, 16B granules ----
template<int N>
__device__ __forceinline__ void wait_async()
{
    asm volatile("s_wait_asynccnt %0" :: "n"(N));
}

template<int ITER, int GR>
__device__ __forceinline__ void copy_chunk(const __bf16* src, __bf16* dstLds, int tid)
{
    unsigned base = (unsigned)(uintptr_t)dstLds;   // LDS-aperture generic ptr: low 32 bits = LDS offset
    #pragma unroll
    for (int it = 0; it < ITER; ++it) {
        int i = tid + it * NTHD;
        if (i >= GR) i -= GR;                      // wrap => uniform per-thread issue count
        unsigned lds = base + (unsigned)i * 16u;
        const void* g = (const char*)src + (size_t)i * 16;
        asm volatile("global_load_async_to_lds_b128 %0, %1, off"
                     :: "v"(lds), "v"(g) : "memory");
    }
}

// ---------------- grid-wide barrier (all NBLK blocks resident) ----------------
__device__ __forceinline__ void grid_sync(unsigned* cnt, unsigned* gen, unsigned& done)
{
    __syncthreads();
    if (threadIdx.x == 0) {
        __threadfence();
        unsigned prev = __hip_atomic_fetch_add(cnt, 1u, __ATOMIC_ACQ_REL, __HIP_MEMORY_SCOPE_AGENT);
        if (prev == (unsigned)(NBLK - 1)) {
            __hip_atomic_store(cnt, 0u, __ATOMIC_RELAXED, __HIP_MEMORY_SCOPE_AGENT);
            __hip_atomic_fetch_add(gen, 1u, __ATOMIC_RELEASE, __HIP_MEMORY_SCOPE_AGENT);
        } else {
            while (__hip_atomic_load(gen, __ATOMIC_ACQUIRE, __HIP_MEMORY_SCOPE_AGENT) < done + 1u)
                __builtin_amdgcn_s_sleep(1);
        }
        __threadfence();
    }
    done++;
    __syncthreads();
}

// ---------------- double-buffered async GEMM ----------------
// acc[32 x NT*16] = Ab(bf16,LDS,[32][KP]) @ W (bf16, pre-transposed chunk-major:
//   Wb[kc][n*32+kk] = W[kc*32+kk][n], zero-padded).
template<int NT, int KC>
__device__ __forceinline__ void gemm_async(v8f* acc, const __bf16* Ab,
                                           __bf16* Wt0, __bf16* Wt1,
                                           const __bf16* Wb,
                                           int wave, int lane, int tid)
{
    constexpr int NTILES = 2 * NT;                 // 2 row tiles x NT col tiles
    constexpr int TPW    = (NTILES + 7) / 8;
    constexpr int NC     = NT * 16;                // 272 or 256
    constexpr int GR     = (NC * 32 * 2) / 16;     // 16B granules per chunk
    constexpr int ITER   = (GR + NTHD - 1) / NTHD; // async ops per thread per chunk
    #pragma unroll
    for (int ti = 0; ti < TPW; ++ti)
        #pragma unroll
        for (int j = 0; j < 8; ++j) acc[ti][j] = 0.f;

    const int nco = lane & 15;
    const int hi  = lane >> 4;

    copy_chunk<ITER, GR>(Wb, Wt0, tid);            // prime buffer 0
    for (int kc = 0; kc < KC; ++kc) {
        __bf16* Wcur = (kc & 1) ? Wt1 : Wt0;
        if (kc + 1 < KC) {
            __bf16* Wnxt = (kc & 1) ? Wt0 : Wt1;
            copy_chunk<ITER, GR>(Wb + (size_t)(kc + 1) * NC * 32, Wnxt, tid);
            wait_async<ITER>();                    // chunk kc done; kc+1 still in flight
        } else {
            wait_async<0>();
        }
        __syncthreads();
        const int k0 = kc * 32;
        #pragma unroll
        for (int ti = 0; ti < TPW; ++ti) {
            const int tt = wave + 8 * ti;          // wave-uniform guard: EXEC stays full
            if (tt < NTILES) {
                const int mt = tt & 1, nt = tt >> 1;   // mt wave-constant -> A reuse
                const int m  = mt * 16 + nco;
                const __bf16* ap = Ab + (size_t)m * KP + k0 + hi * 8;
                v8bf alo = *(const v8bf*)ap;
                v8bf ahi = *(const v8bf*)(ap + 16);
                const __bf16* bp = Wcur + (nt * 16 + nco) * 32 + hi * 8;
                v8bf blo = *(const v8bf*)bp;
                v8bf bhi = *(const v8bf*)(bp + 16);
                v16bf af, bfv;
                #pragma unroll
                for (int i = 0; i < 8; ++i) {
                    af[i] = alo[i]; af[i + 8] = ahi[i];
                    bfv[i] = blo[i]; bfv[i + 8] = bhi[i];
                }
                acc[ti] = __builtin_amdgcn_wmma_f32_16x16x32_bf16(
                    false, af, false, bfv, (short)0, acc[ti], false, false);
            }
        }
        __syncthreads();                           // all reads done before buf reuse
    }
}

// ---------------- BN: add bias, accumulate block-local column stats (LDS atomics) ----
template<int NT>
__device__ __forceinline__ void bn_stats(v8f* acc, const float* bias, int Hact,
                                         float* colsum, float* colsq, int wave, int lane)
{
    constexpr int NTILES = 2 * NT;
    constexpr int TPW    = (NTILES + 7) / 8;
    const int nco = lane & 15;
    #pragma unroll
    for (int ti = 0; ti < TPW; ++ti) {
        const int tt = wave + 8 * ti;
        if (tt < NTILES) {
            const int nt = tt >> 1;
            const int n  = nt * 16 + nco;
            const float bv = (n < Hact) ? bias[n] : 0.f;
            float s = 0.f, sq = 0.f;
            #pragma unroll
            for (int j = 0; j < 8; ++j) {
                float y = acc[ti][j] + bv;
                acc[ti][j] = y;
                s += y; sq += y * y;
            }
            atomicAdd(&colsum[n], s);              // ds_add_f32
            atomicAdd(&colsq[n],  sq);
        }
    }
}

// ---------------- BN apply + ReLU, write bf16 activations back into Ab ----------------
template<int NT>
__device__ __forceinline__ void bn_apply(v8f* acc, const float* gs, const float* g,
                                         const float* be, int Hact, __bf16* Ab,
                                         int wave, int lane)
{
    constexpr int NTILES = 2 * NT;
    constexpr int TPW    = (NTILES + 7) / 8;
    const int nco = lane & 15;
    const int hi  = lane >> 4;
    #pragma unroll
    for (int ti = 0; ti < TPW; ++ti) {
        const int tt = wave + 8 * ti;
        if (tt < NTILES) {
            const int mt = tt & 1, nt = tt >> 1;
            const int n  = nt * 16 + nco;
            float mu = 0.f, sc = 0.f, sh = 0.f;
            if (n < Hact) {
                float su = gs[n], sq = gs[HP + n];
                mu = su * (1.f / NB);
                float var = sq * (1.f / NB) - mu * mu;
                sc = g[n] * rsqrtf(var + EPSBN);
                sh = be[n];
            }
            const int mbase = mt * 16 + hi * 8;
            #pragma unroll
            for (int j = 0; j < 8; ++j) {
                float y = (acc[ti][j] - mu) * sc + sh;   // 0 for padded cols
                y = fmaxf(y, 0.f);
                Ab[(mbase + j) * KP + n] = (__bf16)y;
            }
        }
    }
}

// ---------------- init kernel: zero barrier + per-event BN stat slots ----------------
__global__ void init_ws(unsigned* cnt, unsigned* gen, float* gstats, unsigned nstats)
{
    unsigned i = blockIdx.x * blockDim.x + threadIdx.x;
    if (i == 0) { *cnt = 0u; *gen = 0u; }
    for (; i < nstats; i += gridDim.x * blockDim.x) gstats[i] = 0.f;
}

// ---------------- one-time weight conversion: f32 [K][H] -> bf16 chunk-major W^T ----
__device__ __forceinline__ void conv_group(const float* src, __bf16* dst,
                                           int inst, int Kact, int Hact, int KC, int NC,
                                           size_t gid, size_t gsz)
{
    const size_t total     = (size_t)inst * KC * 32 * NC;
    const size_t srcStride = (size_t)Kact * Hact;
    const size_t dstStride = (size_t)KC * NC * 32;
    for (size_t idx = gid; idx < total; idx += gsz) {
        int    n  = (int)(idx % NC);           // fastest: coalesced src reads
        size_t r  = idx / NC;
        int    kk = (int)(r & 31);
        size_t r2 = r >> 5;
        int    kc = (int)(r2 % KC);
        int    t  = (int)(r2 / KC);
        int    kg = kc * 32 + kk;
        float  v  = (kg < Kact && n < Hact) ? src[(size_t)t * srcStride + (size_t)kg * Hact + n] : 0.f;
        dst[(size_t)t * dstStride + (size_t)kc * NC * 32 + (size_t)n * 32 + kk] = (__bf16)v;
    }
}

__global__ void convert_weights(const float* W1, const float* W2, const float* W3,
                                const float* Wv1, const float* Wv2,
                                __bf16* wb1, __bf16* wb2, __bf16* wb3,
                                __bf16* wbv1, __bf16* wbv2)
{
    const size_t gsz = (size_t)gridDim.x * blockDim.x;
    const size_t gid = (size_t)blockIdx.x * blockDim.x + threadIdx.x;
    conv_group(W1,  wb1,  SS, DD, HH, 8, HP,  gid, gsz);   // D->H
    conv_group(W2,  wb2,  SS, HH, HH, 9, HP,  gid, gsz);   // H->H
    conv_group(W3,  wb3,  SS, HH, DD, 9, 256, gid, gsz);   // H->D
    conv_group(Wv1, wbv1, 1,  DD, HH, 8, HP,  gid, gsz);
    conv_group(Wv2, wbv2, 1,  HH, HH, 9, HP,  gid, gsz);
}

// ---------------- persistent fused scan kernel ----------------
__global__ __launch_bounds__(NTHD)
void deepbsde_persistent(
    const float* __restrict__ x,  const float* __restrict__ dW,
    const float* __restrict__ law, const float* __restrict__ tg,
    const __bf16* __restrict__ wb1, const float* __restrict__ b1,
    const float* __restrict__ g1, const float* __restrict__ be1,
    const __bf16* __restrict__ wb2, const float* __restrict__ b2,
    const float* __restrict__ g2, const float* __restrict__ be2,
    const __bf16* __restrict__ wb3, const float* __restrict__ b3,
    const __bf16* __restrict__ wbv1, const float* __restrict__ bv1,
    const float* __restrict__ gv1, const float* __restrict__ bev1,
    const __bf16* __restrict__ wbv2, const float* __restrict__ bv2,
    const float* __restrict__ gv2, const float* __restrict__ bev2,
    const float* __restrict__ Wv3, const float* __restrict__ bv3,
    const float* __restrict__ gv3, const float* __restrict__ bev3,
    float* __restrict__ out, float* __restrict__ gstats, float* __restrict__ xcur,
    unsigned* cnt, unsigned* gen)
{
    __shared__ __bf16 Ab[ROWS * KP];      // 18.0 KB activations (bf16)
    __shared__ __bf16 Wt0[HP * 32];       // 17.0 KB W^T chunk buffer 0
    __shared__ __bf16 Wt1[HP * 32];       // 17.0 KB W^T chunk buffer 1
    __shared__ float  colsum[HP], colsq[HP];
    __shared__ float  rowsq[ROWS], rowxl[ROWS], rowdot[ROWS], rowy[ROWS], vloc[ROWS];

    const int tid  = threadIdx.x;
    const int wave = tid >> 5;
    const int lane = tid & 31;
    const int row0 = blockIdx.x * ROWS;
    unsigned  done = 0;
    v8f acc[5];
    int evt = 0;

    // ---- load x into Ab (bf16) and xcur (f32 master copy); block-local rows only ----
    for (int idx = tid; idx < ROWS * KP; idx += NTHD) {
        int r = idx / KP, c = idx - r * KP;
        float v = 0.f;
        if (c < DD) {
            v = x[(size_t)(row0 + r) * DD + c];
            xcur[(size_t)(row0 + r) * DD + c] = v;
        }
        Ab[idx] = (__bf16)v;
    }

    // ================= v0 network =================
    // L1: D -> H
    for (int i = tid; i < HP; i += NTHD) { colsum[i] = 0.f; colsq[i] = 0.f; }
    gemm_async<17, 8>(acc, Ab, Wt0, Wt1, wbv1, wave, lane, tid);
    bn_stats<17>(acc, bv1, HH, colsum, colsq, wave, lane);
    __syncthreads();
    { float* gs = gstats + (size_t)evt * 2 * HP;
      for (int i = tid; i < HP; i += NTHD) { atomicAdd(&gs[i], colsum[i]); atomicAdd(&gs[HP + i], colsq[i]); } }
    grid_sync(cnt, gen, done);
    bn_apply<17>(acc, gstats + (size_t)evt * 2 * HP, gv1, bev1, HH, Ab, wave, lane);
    evt++;

    // L2: H -> H
    for (int i = tid; i < HP; i += NTHD) { colsum[i] = 0.f; colsq[i] = 0.f; }
    gemm_async<17, 9>(acc, Ab, Wt0, Wt1, wbv2, wave, lane, tid);
    bn_stats<17>(acc, bv2, HH, colsum, colsq, wave, lane);
    __syncthreads();
    { float* gs = gstats + (size_t)evt * 2 * HP;
      for (int i = tid; i < HP; i += NTHD) { atomicAdd(&gs[i], colsum[i]); atomicAdd(&gs[HP + i], colsq[i]); } }
    grid_sync(cnt, gen, done);
    bn_apply<17>(acc, gstats + (size_t)evt * 2 * HP, gv2, bev2, HH, Ab, wave, lane);
    evt++;

    // L3: H -> 1 (per-row dot, then batch BN + ReLU)
    __syncthreads();
    if (tid < ROWS) {
        float dot = bv3[0];
        for (int k = 0; k < HH; ++k) dot += (float)Ab[tid * KP + k] * Wv3[k];
        rowy[tid] = dot;
    }
    __syncthreads();
    if (tid == 0) {
        float s = 0.f, sq = 0.f;
        for (int r = 0; r < ROWS; ++r) { float y = rowy[r]; s += y; sq += y * y; }
        float* gs = gstats + (size_t)evt * 2 * HP;
        atomicAdd(&gs[0], s); atomicAdd(&gs[HP], sq);
    }
    grid_sync(cnt, gen, done);
    {
        const float* gs = gstats + (size_t)evt * 2 * HP;
        float mu  = gs[0] * (1.f / NB);
        float var = gs[HP] * (1.f / NB) - mu * mu;
        float sc  = gv3[0] * rsqrtf(var + EPSBN);
        float sh  = bev3[0];
        if (tid < ROWS) vloc[tid] = fmaxf((rowy[tid] - mu) * sc + sh, 0.f);
    }
    evt++;
    __syncthreads();

    // ---- restore Ab = x (v0 net overwrote it) ----
    for (int idx = tid; idx < ROWS * KP; idx += NTHD) {
        int r = idx / KP, c = idx - r * KP;
        float v = (c < DD) ? xcur[(size_t)(row0 + r) * DD + c] : 0.f;
        Ab[idx] = (__bf16)v;
    }

    // ================= time-step scan =================
    for (int t = 0; t < SS; ++t) {
        const float hstep = tg[t + 1] - tg[t];
        const float snh   = SIGMA * sqrtf(hstep);
        const __bf16* wb1t = wb1 + (size_t)t * 8 * HP * 32;
        const __bf16* wb2t = wb2 + (size_t)t * 9 * HP * 32;
        const __bf16* wb3t = wb3 + (size_t)t * 9 * 256 * 32;
        const float* b1t = b1 + (size_t)t * HH;  const float* g1t = g1 + (size_t)t * HH;
        const float* be1t = be1 + (size_t)t * HH;
        const float* b2t = b2 + (size_t)t * HH;  const float* g2t = g2 + (size_t)t * HH;
        const float* be2t = be2 + (size_t)t * HH;
        const float* b3t = b3 + (size_t)t * DD;
        const float* lawt = law + (size_t)t * DD;
        const float* dWt  = dW + (size_t)t * NB * DD;

        // ---- L1: x @ W1 + b1 -> BN -> ReLU ----
        for (int i = tid; i < HP; i += NTHD) { colsum[i] = 0.f; colsq[i] = 0.f; }
        gemm_async<17, 8>(acc, Ab, Wt0, Wt1, wb1t, wave, lane, tid);
        bn_stats<17>(acc, b1t, HH, colsum, colsq, wave, lane);
        __syncthreads();
        { float* gs = gstats + (size_t)evt * 2 * HP;
          for (int i = tid; i < HP; i += NTHD) { atomicAdd(&gs[i], colsum[i]); atomicAdd(&gs[HP + i], colsq[i]); } }
        grid_sync(cnt, gen, done);
        bn_apply<17>(acc, gstats + (size_t)evt * 2 * HP, g1t, be1t, HH, Ab, wave, lane);
        evt++;

        // ---- L2: h1 @ W2 + b2 -> BN -> ReLU ----
        for (int i = tid; i < HP; i += NTHD) { colsum[i] = 0.f; colsq[i] = 0.f; }
        gemm_async<17, 9>(acc, Ab, Wt0, Wt1, wb2t, wave, lane, tid);
        bn_stats<17>(acc, b2t, HH, colsum, colsq, wave, lane);
        __syncthreads();
        { float* gs = gstats + (size_t)evt * 2 * HP;
          for (int i = tid; i < HP; i += NTHD) { atomicAdd(&gs[i], colsum[i]); atomicAdd(&gs[HP + i], colsq[i]); } }
        grid_sync(cnt, gen, done);
        bn_apply<17>(acc, gstats + (size_t)evt * 2 * HP, g2t, be2t, HH, Ab, wave, lane);
        evt++;

        // ---- L3: grad = h2 @ W3 + b3 (no BN); fused f / noise / x,v update ----
        if (tid < ROWS) { rowsq[tid] = 0.f; rowxl[tid] = 0.f; rowdot[tid] = 0.f; }
        gemm_async<16, 9>(acc, Ab, Wt0, Wt1, wb3t, wave, lane, tid);
        {
            const int nco = lane & 15;
            const int hi  = lane >> 4;
            #pragma unroll
            for (int ti = 0; ti < 4; ++ti) {
                const int tt = wave + 8 * ti;      // always < 32
                const int mt = tt & 1, nt = tt >> 1;
                const int n  = nt * 16 + nco;      // n < 256
                const float b3v = b3t[n];
                const float lw  = lawt[n];
                const int mbase = mt * 16 + hi * 8;
                #pragma unroll
                for (int j = 0; j < 8; ++j) {
                    const int rl = mbase + j;
                    const size_t gidx = (size_t)(row0 + rl) * DD + n;
                    float gr = acc[ti][j] + b3v;
                    float xv = xcur[gidx];
                    float dl = xv - lw;
                    float nz = snh * dWt[gidx];
                    atomicAdd(&rowsq[rl],  gr * gr);
                    atomicAdd(&rowxl[rl],  dl * dl);
                    atomicAdd(&rowdot[rl], gr * nz);
                    float xn = xv - gr * hstep + nz;
                    xcur[gidx] = xn;
                    Ab[rl * KP + n] = (__bf16)xn;
                }
            }
        }
        __syncthreads();
        if (tid < ROWS) {
            float f = 0.5f * (KAPPA * KAPPA) * rowxl[tid] + 0.5f * rowsq[tid];
            vloc[tid] += rowdot[tid] - f * hstep;
        }
        __syncthreads();
    }

    if (tid < ROWS) out[row0 + tid] = vloc[tid];
}

// ---------------- harness entry ----------------
extern "C" void kernel_launch(void* const* d_in, const int* in_sizes, int n_in,
                              void* d_out, int out_size, void* d_ws, size_t ws_size,
                              hipStream_t stream)
{
    (void)in_sizes; (void)n_in; (void)out_size; (void)ws_size;
    const float* x    = (const float*)d_in[0];
    const float* dW   = (const float*)d_in[1];
    const float* law  = (const float*)d_in[2];
    const float* tg   = (const float*)d_in[3];
    const float* W1   = (const float*)d_in[4];
    const float* b1   = (const float*)d_in[5];
    const float* g1   = (const float*)d_in[6];
    const float* be1  = (const float*)d_in[7];
    const float* W2   = (const float*)d_in[8];
    const float* b2   = (const float*)d_in[9];
    const float* g2   = (const float*)d_in[10];
    const float* be2  = (const float*)d_in[11];
    const float* W3   = (const float*)d_in[12];
    const float* b3   = (const float*)d_in[13];
    const float* Wv1  = (const float*)d_in[14];
    const float* bv1  = (const float*)d_in[15];
    const float* gv1  = (const float*)d_in[16];
    const float* bev1 = (const float*)d_in[17];
    const float* Wv2  = (const float*)d_in[18];
    const float* bv2  = (const float*)d_in[19];
    const float* gv2  = (const float*)d_in[20];
    const float* bev2 = (const float*)d_in[21];
    const float* Wv3  = (const float*)d_in[22];
    const float* bv3  = (const float*)d_in[23];
    const float* gv3  = (const float*)d_in[24];
    const float* bev3 = (const float*)d_in[25];

    // ---- workspace layout ----
    char* ws = (char*)d_ws;
    unsigned* cnt    = (unsigned*)ws;
    unsigned* gen    = cnt + 1;
    float*    gstats = (float*)(ws + 256);                       // 103*2*272*4 = 224,128 B
    float*    xcur   = (float*)(ws + (size_t)(256u << 10));      // 2 MB
    size_t woff = (size_t)(256u << 10) + (size_t)NB * DD * 4;    // 2.25 MB
    __bf16* wb1  = (__bf16*)(ws + woff); woff += (size_t)SS * 8 * HP  * 32 * 2;  // 6.96 MB
    __bf16* wb2  = (__bf16*)(ws + woff); woff += (size_t)SS * 9 * HP  * 32 * 2;  // 7.83 MB
    __bf16* wb3  = (__bf16*)(ws + woff); woff += (size_t)SS * 9 * 256 * 32 * 2;  // 7.37 MB
    __bf16* wbv1 = (__bf16*)(ws + woff); woff += (size_t)8 * HP * 32 * 2;
    __bf16* wbv2 = (__bf16*)(ws + woff); woff += (size_t)9 * HP * 32 * 2;

    const unsigned nstats = (unsigned)NEVT * 2u * HP;
    init_ws<<<64, 256, 0, stream>>>(cnt, gen, gstats, nstats);
    convert_weights<<<1024, 256, 0, stream>>>(W1, W2, W3, Wv1, Wv2,
                                              wb1, wb2, wb3, wbv1, wbv2);
    deepbsde_persistent<<<NBLK, NTHD, 0, stream>>>(
        x, dW, law, tg,
        wb1, b1, g1, be1, wb2, b2, g2, be2, wb3, b3,
        wbv1, bv1, gv1, bev1, wbv2, bv2, gv2, bev2, Wv3, bv3, gv3, bev3,
        (float*)d_out, gstats, xcur, cnt, gen);
}
